// VarPatch_33483565039865
// MI455X (gfx1250) — matrix operand order repaired and checked by reading, not compiled, not run on hardware
//
#include <hip/hip_runtime.h>
#include <hip/hip_bf16.h>

// ---------------------------------------------------------------------------
// VarPatch (Mamba + FFN) for MI455X / gfx1250, wave32, WMMA f16->f32,
// async global->LDS staging (ASYNCcnt) with double buffering.
// ---------------------------------------------------------------------------

typedef _Float16 h8   __attribute__((ext_vector_type(8)));
typedef _Float16 v16h __attribute__((ext_vector_type(16)));
typedef float    v8f  __attribute__((ext_vector_type(8)));

constexpr int M_TOK = 32768;   // B*L = 16*2048 tokens

enum { EPI_INPROJ = 0, EPI_F32 = 1, EPI_RES = 2, EPI_RELU = 3, EPI_ADD = 4 };

// ---------------------------------------------------------------------------
// Generic WMMA GEMM: C[m,n] = sum_k A[m,k] * W[n,k]  (both K-contiguous, f16)
// Block: 128 threads = 4 waves; block tile 64x64; wave tile 32x32 (2x2 frags);
// K-step 32; global->LDS via async B128 copies, double-buffered.
// ---------------------------------------------------------------------------
template <int EPI, int K>
__global__ __launch_bounds__(128) void gemm_wmma(
    const _Float16* __restrict__ A, const _Float16* __restrict__ W,
    const float* __restrict__ bias, const float* __restrict__ addsrc,
    float* __restrict__ outF, _Float16* __restrict__ outH0,
    _Float16* __restrict__ outH1, int M, int N) {
  __shared__ _Float16 lA[2][64 * 32];
  __shared__ _Float16 lB[2][64 * 32];
  const int tid = threadIdx.x;
  const int lane = tid & 31;
  const int wave = tid >> 5;
  const int mw = wave >> 1, nw = wave & 1;
  const int m0 = blockIdx.y * 64;
  const int n0 = blockIdx.x * 64;
  const int hi16 = lane >> 4;   // 0 or 1 (half-wave)
  const int r16 = lane & 15;

  // staging geometry: 64 rows x 32 halves = 256 chunks of 8 halves (16B);
  // this thread owns chunks tid and tid+128.
  const int rowC0 = tid >> 2, colC0 = (tid & 3) * 8;
  const int rowC1 = (tid + 128) >> 2, colC1 = ((tid + 128) & 3) * 8;
  const unsigned ldsOffA0 = (unsigned)(rowC0 * 32 + colC0) * 2;
  const unsigned ldsOffA1 = (unsigned)(rowC1 * 32 + colC1) * 2;
  const unsigned long long Abase = (unsigned long long)(uintptr_t)A;
  const unsigned long long Wbase = (unsigned long long)(uintptr_t)W;

  auto issue_tile = [&](int buf, int k0) {
    unsigned la0 = (unsigned)(uintptr_t)(void*)&lA[buf][0] + ldsOffA0;
    unsigned la1 = (unsigned)(uintptr_t)(void*)&lA[buf][0] + ldsOffA1;
    unsigned lb0 = (unsigned)(uintptr_t)(void*)&lB[buf][0] + ldsOffA0;
    unsigned lb1 = (unsigned)(uintptr_t)(void*)&lB[buf][0] + ldsOffA1;
    int ga0 = (int)(((size_t)(m0 + rowC0) * K + k0 + colC0) * 2);
    int ga1 = (int)(((size_t)(m0 + rowC1) * K + k0 + colC1) * 2);
    int gb0 = (int)(((size_t)(n0 + rowC0) * K + k0 + colC0) * 2);
    int gb1 = (int)(((size_t)(n0 + rowC1) * K + k0 + colC1) * 2);
    asm volatile("global_load_async_to_lds_b128 %0, %1, %2"
                 :: "v"(la0), "v"(ga0), "s"(Abase) : "memory");
    asm volatile("global_load_async_to_lds_b128 %0, %1, %2"
                 :: "v"(la1), "v"(ga1), "s"(Abase) : "memory");
    asm volatile("global_load_async_to_lds_b128 %0, %1, %2"
                 :: "v"(lb0), "v"(gb0), "s"(Wbase) : "memory");
    asm volatile("global_load_async_to_lds_b128 %0, %1, %2"
                 :: "v"(lb1), "v"(gb1), "s"(Wbase) : "memory");
  };

  v8f acc[2][2] = {};
  constexpr int NK = K / 32;

  issue_tile(0, 0);
  for (int kt = 0; kt < NK; ++kt) {
    const int cur = kt & 1;
    if (kt + 1 < NK) {
      issue_tile(1 - cur, (kt + 1) * 32);
      asm volatile("s_wait_asynccnt 4" ::: "memory");  // tile kt complete
    } else {
      asm volatile("s_wait_asynccnt 0" ::: "memory");
    }
    __syncthreads();

    // A fragment: lane<16 -> row M=r16, K {0..7,16..23}; lane>=16 -> K {8..15,24..31}
    v16h afrag[2], bfrag[2];
#pragma unroll
    for (int i = 0; i < 2; ++i) {
      const _Float16* pa = &lA[cur][(mw * 32 + i * 16 + r16) * 32 + hi16 * 8];
      union { v16h v; h8 h[2]; } u;
      u.h[0] = *(const h8*)pa;
      u.h[1] = *(const h8*)(pa + 16);
      afrag[i] = u.v;
    }
    // B fragment: lane<16 -> col N=r16, K {0..7,16..23}; lane>=16 -> K {8..15,24..31}
#pragma unroll
    for (int j = 0; j < 2; ++j) {
      const _Float16* pb = &lB[cur][(nw * 32 + j * 16 + r16) * 32 + hi16 * 8];
      union { v16h v; h8 h[2]; } u;
      u.h[0] = *(const h8*)pb;
      u.h[1] = *(const h8*)(pb + 16);
      bfrag[j] = u.v;
    }
#pragma unroll
    for (int i = 0; i < 2; ++i)
#pragma unroll
      for (int j = 0; j < 2; ++j)
        acc[i][j] = __builtin_amdgcn_wmma_f32_16x16x32_f16(
            false, afrag[i], false, bfrag[j], (short)0, acc[i][j], false, false);
    __syncthreads();  // all waves done reading buf `cur` (reused at kt+2)
  }

  // Epilogue. C/D layout: VGPR r, lane: M = r + 8*(lane>=16), N = lane&15.
#pragma unroll
  for (int i = 0; i < 2; ++i) {
#pragma unroll
    for (int j = 0; j < 2; ++j) {
      const int nn = n0 + nw * 32 + j * 16 + r16;
#pragma unroll
      for (int r = 0; r < 8; ++r) {
        const int mm = m0 + mw * 32 + i * 16 + r + hi16 * 8;
        float v = acc[i][j][r];
        if (EPI == EPI_INPROJ) {
          v += bias[nn];
          if (nn < 512) {
            outH0[(size_t)mm * 512 + nn] = (_Float16)v;          // xi (pre-conv)
          } else {
            float s = v / (1.f + __expf(-v));                     // silu(z)
            outH1[(size_t)mm * 512 + (nn - 512)] = (_Float16)s;
          }
        } else if (EPI == EPI_F32) {
          if (nn < N) outF[(size_t)mm * N + nn] = v;              // x_dbl
        } else if (EPI == EPI_RES) {
          outF[(size_t)mm * N + nn] = v + addsrc[(size_t)mm * N + nn];
        } else if (EPI == EPI_RELU) {
          v += bias[nn];
          v = v > 0.f ? v : 0.f;
          outH0[(size_t)mm * N + nn] = (_Float16)v;
        } else {  // EPI_ADD: bias + residual
          outF[(size_t)mm * N + nn] =
              v + bias[nn] + addsrc[(size_t)mm * N + nn];
        }
      }
    }
  }
}

// ---------------------------------------------------------------------------
// f32 -> f16 convert
// ---------------------------------------------------------------------------
__global__ void f32_to_f16(const float* __restrict__ in,
                           _Float16* __restrict__ out, int n) {
  int i = blockIdx.x * 256 + threadIdx.x;
  if (i < n) out[i] = (_Float16)in[i];
}

// ---------------------------------------------------------------------------
// Causal depthwise conv1d (k=4, left pad 3) along L per batch + SiLU.
// ---------------------------------------------------------------------------
__global__ __launch_bounds__(256) void conv_silu(
    const _Float16* __restrict__ xiraw, const float* __restrict__ cw,
    const float* __restrict__ cb, _Float16* __restrict__ xi16) {
  size_t idx = (size_t)blockIdx.x * 256 + threadIdx.x;  // over M_TOK*512
  int c = (int)(idx & 511);
  size_t m = idx >> 9;
  int l = (int)(m & 2047);
  float acc = cb[c];
#pragma unroll
  for (int k = 0; k < 4; ++k) {
    int ls = l - 3 + k;
    if (ls >= 0)
      acc += cw[c * 4 + k] * (float)xiraw[(m + (size_t)(ls - l)) * 512 + c];
  }
  float s = acc / (1.f + __expf(-acc));
  xi16[idx] = (_Float16)s;
}

// ---------------------------------------------------------------------------
// Selective scan. One block per batch b (512 threads = one d-channel each).
// Fuses dt_proj + softplus, SSM recurrence, Dp skip, and silu(z) gating.
// Emits the gated activation in f16 (out_proj GEMM operand).
// ---------------------------------------------------------------------------
__global__ __launch_bounds__(512) void mamba_scan(
    const float* __restrict__ xdbl, const _Float16* __restrict__ xi16,
    const _Float16* __restrict__ sz16, const float* __restrict__ dtw,
    const float* __restrict__ dtb, const float* __restrict__ A_log,
    const float* __restrict__ Dp, _Float16* __restrict__ yact) {
  __shared__ float sdbl[32 * 48];
  const int d = threadIdx.x;
  const int b = blockIdx.x;
  float Arow[16], Wrow[16], h[16];
#pragma unroll
  for (int n = 0; n < 16; ++n) {
    Arow[n] = -__expf(A_log[d * 16 + n]);
    Wrow[n] = dtw[d * 16 + n];
    h[n] = 0.f;
  }
  const float db = dtb[d];
  const float Dpd = Dp[d];

  for (int l0 = 0; l0 < 2048; l0 += 32) {
    __syncthreads();
#pragma unroll
    for (int t = 0; t < 3; ++t) {
      int idx = d + t * 512;
      sdbl[idx] = xdbl[(size_t)(b * 2048 + l0) * 48 + idx];
    }
    __syncthreads();
    for (int li = 0; li < 32; ++li) {
      const float* row = &sdbl[li * 48];
      float pre = db;
#pragma unroll
      for (int r = 0; r < 16; ++r) pre += row[r] * Wrow[r];
      float delta = pre > 20.f ? pre : log1pf(__expf(pre));  // softplus
      size_t tok = (size_t)(b * 2048 + l0 + li);
      float u = (float)xi16[tok * 512 + d];
      float du = delta * u;
      float y = 0.f;
#pragma unroll
      for (int n = 0; n < 16; ++n) {
        float dA = __expf(delta * Arow[n]);
        h[n] = dA * h[n] + du * row[16 + n];
        y += h[n] * row[32 + n];
      }
      float yo = y + u * Dpd;
      float g = (float)sz16[tok * 512 + d];
      yact[tok * 512 + d] = (_Float16)(yo * g);
    }
  }
}

// ---------------------------------------------------------------------------
// LayerNorm over rows of 256; one wave per row; emits f32 + f16.
// ---------------------------------------------------------------------------
__global__ __launch_bounds__(256) void ln1_kernel(
    const float* __restrict__ in, const float* __restrict__ g,
    const float* __restrict__ bta, float* __restrict__ outF,
    _Float16* __restrict__ outH) {
  const int wave = threadIdx.x >> 5, lane = threadIdx.x & 31;
  const int row = blockIdx.x * 8 + wave;
  const float* p = in + (size_t)row * 256;
  float v[8];
  float s = 0.f;
#pragma unroll
  for (int i = 0; i < 8; ++i) {
    v[i] = p[lane + 32 * i];
    s += v[i];
  }
#pragma unroll
  for (int o = 16; o > 0; o >>= 1) s += __shfl_xor(s, o, 32);
  float mu = s * (1.f / 256.f);
  float ss = 0.f;
#pragma unroll
  for (int i = 0; i < 8; ++i) {
    float dv = v[i] - mu;
    ss += dv * dv;
  }
#pragma unroll
  for (int o = 16; o > 0; o >>= 1) ss += __shfl_xor(ss, o, 32);
  float inv = rsqrtf(ss * (1.f / 256.f) + 1e-5f);
#pragma unroll
  for (int i = 0; i < 8; ++i) {
    int c = lane + 32 * i;
    float o2 = (v[i] - mu) * inv * g[c] + bta[c];
    outF[(size_t)row * 256 + c] = o2;
    outH[(size_t)row * 256 + c] = (_Float16)o2;
  }
}

// ---------------------------------------------------------------------------
// Tail: LN(norm2) -> LN(ln) -> transpose (p,d)->(d,p) via LDS -> + x residual.
// One block per (b, nvar): 64 rows (patches) x 256 cols, in two half-slabs.
// ---------------------------------------------------------------------------
__global__ __launch_bounds__(256) void final_kernel(
    const float* __restrict__ tmp2, const float* __restrict__ g2,
    const float* __restrict__ b2, const float* __restrict__ gl,
    const float* __restrict__ bl, const float* __restrict__ xin,
    float* __restrict__ out) {
  __shared__ float tile[32 * 257];
  const int bv = blockIdx.x;  // 0..511
  const int wave = threadIdx.x >> 5, lane = threadIdx.x & 31;

  for (int half = 0; half < 2; ++half) {
    const int pbase = half * 32;
#pragma unroll
    for (int i = 0; i < 4; ++i) {
      int p = pbase + wave * 4 + i;
      const float* rowp = tmp2 + ((size_t)bv * 64 + p) * 256;
      float v[8];
      float s = 0.f;
#pragma unroll
      for (int j = 0; j < 8; ++j) {
        v[j] = rowp[lane + 32 * j];
        s += v[j];
      }
#pragma unroll
      for (int o = 16; o > 0; o >>= 1) s += __shfl_xor(s, o, 32);
      float mu = s * (1.f / 256.f);
      float ss = 0.f;
#pragma unroll
      for (int j = 0; j < 8; ++j) {
        float dv = v[j] - mu;
        ss += dv * dv;
      }
#pragma unroll
      for (int o = 16; o > 0; o >>= 1) ss += __shfl_xor(ss, o, 32);
      float inv = rsqrtf(ss * (1.f / 256.f) + 1e-5f);
#pragma unroll
      for (int j = 0; j < 8; ++j) {
        int c = lane + 32 * j;
        v[j] = (v[j] - mu) * inv * g2[c] + b2[c];
      }
      // second LN (ln_g / ln_b)
      s = 0.f;
#pragma unroll
      for (int j = 0; j < 8; ++j) s += v[j];
#pragma unroll
      for (int o = 16; o > 0; o >>= 1) s += __shfl_xor(s, o, 32);
      mu = s * (1.f / 256.f);
      ss = 0.f;
#pragma unroll
      for (int j = 0; j < 8; ++j) {
        float dv = v[j] - mu;
        ss += dv * dv;
      }
#pragma unroll
      for (int o = 16; o > 0; o >>= 1) ss += __shfl_xor(ss, o, 32);
      inv = rsqrtf(ss * (1.f / 256.f) + 1e-5f);
#pragma unroll
      for (int j = 0; j < 8; ++j) {
        int c = lane + 32 * j;
        tile[(p - pbase) * 257 + c] = (v[j] - mu) * inv * gl[c] + bl[c];
      }
    }
    __syncthreads();
    // write transposed: out[bv][d][p] = tile[p][d] + x[bv][d][p]
    {
      const int p = pbase + (threadIdx.x & 31);
      const int dg = threadIdx.x >> 5;  // 0..7
      for (int i = 0; i < 32; ++i) {
        int dd = dg * 32 + i;
        size_t o = ((size_t)bv * 256 + dd) * 64 + p;
        out[o] = tile[(p - pbase) * 257 + dd] + xin[o];
      }
    }
    __syncthreads();
  }
}

// ---------------------------------------------------------------------------
// Host launcher
// ---------------------------------------------------------------------------
extern "C" void kernel_launch(void* const* d_in, const int* in_sizes, int n_in,
                              void* d_out, int out_size, void* d_ws,
                              size_t ws_size, hipStream_t stream) {
  (void)in_sizes; (void)n_in; (void)out_size; (void)ws_size;

  const float* x         = (const float*)d_in[0];   // (16,32,256,64) == h flat
  const float* in_proj_w = (const float*)d_in[1];   // (1024,256)
  const float* in_proj_b = (const float*)d_in[2];   // (1024)
  const float* conv_w    = (const float*)d_in[3];   // (512,4)
  const float* conv_b    = (const float*)d_in[4];   // (512)
  const float* x_proj_w  = (const float*)d_in[5];   // (48,512)
  const float* dt_proj_w = (const float*)d_in[6];   // (512,16)
  const float* dt_proj_b = (const float*)d_in[7];   // (512)
  const float* A_log     = (const float*)d_in[8];   // (512,16)
  const float* Dp        = (const float*)d_in[9];   // (512)
  const float* out_proj_w= (const float*)d_in[10];  // (256,512)
  const float* norm1_g   = (const float*)d_in[11];
  const float* norm1_b   = (const float*)d_in[12];
  const float* ffn1_w    = (const float*)d_in[13];  // (512,256)
  const float* ffn1_b    = (const float*)d_in[14];
  const float* ffn2_w    = (const float*)d_in[15];  // (256,512)
  const float* ffn2_b    = (const float*)d_in[16];
  const float* norm2_g   = (const float*)d_in[17];
  const float* norm2_b   = (const float*)d_in[18];
  const float* ln_g      = (const float*)d_in[19];
  const float* ln_b      = (const float*)d_in[20];
  float* out = (float*)d_out;

  // workspace carve (256B aligned)
  size_t off = 0;
  auto carve = [&](size_t bytes) -> void* {
    void* p = (char*)d_ws + off;
    off += (bytes + 255) & ~(size_t)255;
    return p;
  };
  _Float16* H16     = (_Float16*)carve((size_t)M_TOK * 256 * 2);
  _Float16* Win16   = (_Float16*)carve((size_t)1024 * 256 * 2);
  _Float16* Wxp16   = (_Float16*)carve((size_t)48 * 512 * 2);
  _Float16* Wop16   = (_Float16*)carve((size_t)256 * 512 * 2);
  _Float16* Wf116   = (_Float16*)carve((size_t)512 * 256 * 2);
  _Float16* Wf216   = (_Float16*)carve((size_t)256 * 512 * 2);
  _Float16* XIRAW16 = (_Float16*)carve((size_t)M_TOK * 512 * 2);  // reused: YACT16
  _Float16* SZ16    = (_Float16*)carve((size_t)M_TOK * 512 * 2);
  _Float16* XI16    = (_Float16*)carve((size_t)M_TOK * 512 * 2);
  float*    XDBL    = (float*)carve((size_t)M_TOK * 48 * 4);
  float*    HRES    = (float*)carve((size_t)M_TOK * 256 * 4);     // reused: TMP2
  float*    H1      = (float*)carve((size_t)M_TOK * 256 * 4);
  _Float16* H1_16   = (_Float16*)carve((size_t)M_TOK * 256 * 2);
  _Float16* F1_16   = (_Float16*)carve((size_t)M_TOK * 512 * 2);
  _Float16* YACT16 = XIRAW16;  // alias (xi_raw dead after conv)
  float*    TMP2   = HRES;     // alias (hres dead after ln1)

  // 1) f16 conversions (input tokens + GEMM weights)
  f32_to_f16<<<(M_TOK * 256 + 255) / 256, 256, 0, stream>>>(x, H16, M_TOK * 256);
  f32_to_f16<<<1024, 256, 0, stream>>>(in_proj_w, Win16, 1024 * 256);
  f32_to_f16<<<96, 256, 0, stream>>>(x_proj_w, Wxp16, 48 * 512);
  f32_to_f16<<<512, 256, 0, stream>>>(out_proj_w, Wop16, 256 * 512);
  f32_to_f16<<<512, 256, 0, stream>>>(ffn1_w, Wf116, 512 * 256);
  f32_to_f16<<<512, 256, 0, stream>>>(ffn2_w, Wf216, 256 * 512);

  // 2) in_proj: (32768x256)@(1024x256)^T -> split xi / silu(z)
  gemm_wmma<EPI_INPROJ, 256><<<dim3(16, 512), 128, 0, stream>>>(
      H16, Win16, in_proj_b, nullptr, nullptr, XIRAW16, SZ16, M_TOK, 1024);

  // 3) depthwise causal conv + silu -> xi16
  conv_silu<<<(M_TOK * 512) / 256, 256, 0, stream>>>(XIRAW16, conv_w, conv_b,
                                                     XI16);

  // 4) x_proj: (32768x512)@(48x512)^T -> x_dbl (dt|B|C)
  gemm_wmma<EPI_F32, 512><<<dim3(1, 512), 128, 0, stream>>>(
      XI16, Wxp16, nullptr, nullptr, XDBL, nullptr, nullptr, M_TOK, 48);

  // 5) selective scan (fused dt_proj/softplus/Dp/silu-gate) -> yact16
  mamba_scan<<<16, 512, 0, stream>>>(XDBL, XI16, SZ16, dt_proj_w, dt_proj_b,
                                     A_log, Dp, YACT16);

  // 6) out_proj + residual(h): -> hres
  gemm_wmma<EPI_RES, 512><<<dim3(4, 512), 128, 0, stream>>>(
      YACT16, Wop16, nullptr, x, HRES, nullptr, nullptr, M_TOK, 256);

  // 7) LN1 -> h1 (f32 + f16)
  ln1_kernel<<<M_TOK / 8, 256, 0, stream>>>(HRES, norm1_g, norm1_b, H1, H1_16);

  // 8) ffn1: relu(h1@W1^T+b1) -> f1 (f16)
  gemm_wmma<EPI_RELU, 256><<<dim3(8, 512), 128, 0, stream>>>(
      H1_16, Wf116, ffn1_b, nullptr, nullptr, F1_16, nullptr, M_TOK, 512);

  // 9) ffn2 + bias + h1 residual -> tmp2
  gemm_wmma<EPI_ADD, 512><<<dim3(4, 512), 128, 0, stream>>>(
      F1_16, Wf216, ffn2_b, H1, TMP2, nullptr, nullptr, M_TOK, 256);

  // 10) LN2 -> LN3 -> transpose -> + x_residual -> out
  final_kernel<<<512, 256, 0, stream>>>(TMP2, norm2_g, norm2_b, ln_g, ln_b, x,
                                        out);
}
